// AttentionBlock_61280593380023
// MI455X (gfx1250) — matrix operand order repaired and checked by reading, not compiled, not run on hardware
//
#include <hip/hip_runtime.h>
#include <hip/hip_bf16.h>

// ---------------------------------------------------------------------------
// AttentionBlock for MI455X (gfx1250, wave32, WMMA f32_16x16x32_f16)
// B=8, C=256, N=1024 (32x32), H=4, D=64
// Pipeline: xpose (x -> xs f16) ; qkv GEMM ; flash attention ; out GEMM.
// Scratch layouts (all chosen so WMMA fragments are 128-bit loads):
//   XT (B,N,C) f16 : A-operand of qkv projection   (k = c contiguous)
//   Q  (B,H,N,D) f16 : B-operand of S^T = K @ Q^T  (k = d contiguous)
//   K  (B,H,N,D) f16 : A-operand of S^T            (k = d contiguous)
//   V  (B,H,D,N) f16 : A-operand of O^T = V^T @ P^T (k = j contiguous)
//   O  (B,N,H*D) f16 : A-operand of out-projection (k contiguous)
// ---------------------------------------------------------------------------

typedef __attribute__((ext_vector_type(16))) _Float16 v16h;
typedef __attribute__((ext_vector_type(8)))  _Float16 v8h;
typedef __attribute__((ext_vector_type(2)))  _Float16 h2;
typedef __attribute__((ext_vector_type(8)))  float    v8f;
typedef __attribute__((ext_vector_type(8)))  int      v8i;

#define NB   8
#define NC   256
#define NN   1024
#define NH   4
#define ND   64
#define QKVO 768

__device__ __forceinline__ v8f wmma_f16(v16h a, v16h b, v8f c) {
  return __builtin_amdgcn_wmma_f32_16x16x32_f16(false, a, false, b,
                                                (short)0, c, false, false);
}
__device__ __forceinline__ v16h cat8(v8h a, v8h b) {
  return __builtin_shufflevector(a, b, 0, 1, 2, 3, 4, 5, 6, 7,
                                 8, 9, 10, 11, 12, 13, 14, 15);
}
// A-operand (16x32 f16): lane = row (lane&15); halves s = K kbase+s,
// kbase = (lane<16)?0:16 (same k order as B).
// B-operand (32x16 f16): lane = column (lane&15); halves s = K kbase+s,
// kbase = (lane<16)?0:16.  C/D: lane = column; row = r + ((lane<16)?0:8).

// ---------------------------------------------------------------------------
// Kernel 0: transpose + f16 convert: xt[b,n,c] = (f16) x[b,c,n].
// 64x64 tiles through LDS; both global sides are 128-bit coalesced.
// ---------------------------------------------------------------------------
__global__ __launch_bounds__(256)
void xpose_kernel(const float* __restrict__ x, _Float16* __restrict__ xt) {
  __shared__ _Float16 sT[64][72];              // 72-pitch: 16B-aligned rows
  const int t  = threadIdx.x;
  const int b  = blockIdx.z;
  const int c0 = blockIdx.y * 64;
  const int n0 = blockIdx.x * 64;
  {
    const int cl = t & 63;                     // c row in tile
    const int sg = (t >> 6) * 16;              // n segment
    const float* src = x + ((size_t)b * NC + c0 + cl) * NN + n0 + sg;
    const v8f f0 = *(const v8f*)(src);
    const v8f f1 = *(const v8f*)(src + 8);
#pragma unroll
    for (int u = 0; u < 8; ++u) sT[sg + u][cl]     = (_Float16)f0[u];
#pragma unroll
    for (int u = 0; u < 8; ++u) sT[sg + 8 + u][cl] = (_Float16)f1[u];
  }
  __syncthreads();
  {
    const int nl = t & 63;                     // n row in tile
    const int sg = (t >> 6) * 16;              // c segment
    const v8h o0 = *(const v8h*)(&sT[nl][sg]);
    const v8h o1 = *(const v8h*)(&sT[nl][sg + 8]);
    _Float16* dst = xt + ((size_t)b * NN + n0 + nl) * NC + c0 + sg;
    *(v8h*)(dst)     = o0;
    *(v8h*)(dst + 8) = o1;
  }
}

// ---------------------------------------------------------------------------
// Kernel 1: QKV projection. Block = 8 waves = 8 m-tiles of one n-tile;
// Wp panel converted to f16 in LDS once ([col][k]); A-frags 2x b128 from xt.
// ---------------------------------------------------------------------------
__global__ __launch_bounds__(256)
void qkv_proj_kernel(const _Float16* __restrict__ xt,
                     const float* __restrict__ Wp,
                     const float* __restrict__ bp,
                     _Float16* __restrict__ qbuf,
                     _Float16* __restrict__ kbuf,
                     _Float16* __restrict__ vbuf) {
  __shared__ _Float16 sB[16 * 256];            // [col][k]  8 KB
  const int tid  = threadIdx.x;
  const int lane = tid & 31;
  const int wave = tid >> 5;
  const int col  = lane & 15;
  const int hi   = lane >> 4;
  const int nt   = blockIdx.x % 48;            // n-tile of 768
  const int mt   = (blockIdx.x / 48) * 8 + wave;
  const int col0 = nt * 16;

  {                                            // stage B panel (f32 -> f16)
    const int c  = tid & 15;
    const int k0 = (tid >> 4) * 16;
    v8h t0, t1;
#pragma unroll
    for (int u = 0; u < 8; ++u) {
      t0[u] = (_Float16)Wp[(k0 + u) * QKVO + col0 + c];
      t1[u] = (_Float16)Wp[(k0 + 8 + u) * QKVO + col0 + c];
    }
    *(v8h*)(sB + c * 256 + k0)     = t0;
    *(v8h*)(sB + c * 256 + k0 + 8) = t1;
  }
  __syncthreads();

  const int m = mt * 16 + col;
  const int abase = hi ? 8 : 0;
  const int kbase = hi ? 16 : 0;
  const _Float16* arow = xt + (size_t)m * NC;
  const _Float16* brow = sB + col * 256;

  v8f acc = {};
#pragma unroll
  for (int k0 = 0; k0 < NC; k0 += 32) {
    const v16h afr = cat8(*(const v8h*)(arow + k0 + kbase),
                          *(const v8h*)(arow + k0 + kbase + 8));
    const v16h bfr = cat8(*(const v8h*)(brow + k0 + kbase),
                          *(const v8h*)(brow + k0 + kbase + 8));
    acc = wmma_f16(afr, bfr, acc);
  }

  // block-uniform scatter select (16-wide tiles never straddle q/k/v bounds)
  const int h   = col0 / 192;
  const int e0  = col0 % 192;
  const int sel = e0 / 64;                     // 0=q 1=k 2=v
  const int d   = (e0 - sel * 64) + col;
  const float bias = bp[col0 + col];
  const int rowbase = mt * 16 + 8 * hi;        // same b for whole tile
  const int bb  = rowbase >> 10;
  const int nn0 = rowbase & (NN - 1);
  const int bh  = bb * NH + h;

  if (sel == 2) {                              // V^T: contiguous along n
    v8h o;
#pragma unroll
    for (int r = 0; r < 8; ++r) o[r] = (_Float16)(acc[r] + bias);
    *(v8h*)(vbuf + ((size_t)bh * ND + d) * NN + nn0) = o;
  } else {
    _Float16* dst = (sel == 0) ? qbuf : kbuf;
    _Float16* base = dst + ((size_t)bh * NN + nn0) * ND + d;
#pragma unroll
    for (int r = 0; r < 8; ++r) base[r * ND] = (_Float16)(acc[r] + bias);
  }
}

// ---------------------------------------------------------------------------
// Kernel 2: streaming attention, transposed (S^T, O^T) so each lane owns one
// query column: per-lane softmax state, 2 shuffles per j-step. K and V tiles
// double-buffered in LDS via global_load_async_to_lds_b128 (ASYNCcnt).
// ---------------------------------------------------------------------------
__global__ __launch_bounds__(256)
void attn_kernel(const _Float16* __restrict__ qbuf,
                 const _Float16* __restrict__ kbuf,   // (B,H,N,D)
                 const _Float16* __restrict__ vbuf,   // (B,H,D,N)
                 _Float16* __restrict__ obuf) {
  __shared__ _Float16 sK[2][32 * 64];   // [j][d]  2 x 4 KB
  __shared__ _Float16 sV[2][64 * 32];   // [d][j]  2 x 4 KB
  const int tid  = threadIdx.x;
  const int lane = tid & 31;
  const int wave = tid >> 5;
  const int col  = lane & 15;
  const int hi   = lane >> 4;
  const int abase = hi ? 8 : 0;
  const int kbase = hi ? 16 : 0;

  const int bh = blockIdx.y;
  const int i0 = blockIdx.x * 128 + wave * 16;
  const _Float16* Q = qbuf + (size_t)bh * NN * ND;
  const _Float16* K = kbuf + (size_t)bh * NN * ND;
  const _Float16* V = vbuf + (size_t)bh * ND * NN;

  // Q^T as B-operand: lane holds query column; k = d contiguous.
  const _Float16* qrow = Q + (i0 + col) * ND;
  const v16h qb0 = cat8(*(const v8h*)(qrow + kbase),
                        *(const v8h*)(qrow + kbase + 8));
  const v16h qb1 = cat8(*(const v8h*)(qrow + 32 + kbase),
                        *(const v8h*)(qrow + 32 + kbase + 8));

  const int vd = tid >> 2, vch = (tid & 3) * 8;   // V staging assignment
  auto stage = [&](int j0, int buf) {
    unsigned kdst = (unsigned)(unsigned long long)(void*)(&sK[buf][tid * 8]);
    unsigned kgo  = (unsigned)((j0 * ND + tid * 8) * sizeof(_Float16));
    asm volatile("global_load_async_to_lds_b128 %0, %1, %2"
                 :: "v"(kdst), "v"(kgo),
                    "s"((unsigned long long)(const void*)K) : "memory");
    unsigned vdst = (unsigned)(unsigned long long)(void*)(&sV[buf][vd * 32 + vch]);
    unsigned vgo  = (unsigned)((vd * NN + j0 + vch) * sizeof(_Float16));
    asm volatile("global_load_async_to_lds_b128 %0, %1, %2"
                 :: "v"(vdst), "v"(vgo),
                    "s"((unsigned long long)(const void*)V) : "memory");
  };

  float mrun = -3.0e38f, lrun = 0.0f;
  v8f acc[4] = {};
  const float scale = 0.125f;                    // D^-0.5

  stage(0, 0);
  int cur = 0;
  for (int j0 = 0; j0 < NN; j0 += 32) {
    asm volatile("s_wait_asynccnt 0" ::: "memory");
    __syncthreads();                             // tile ready, prev reads done
    if (j0 + 32 < NN) stage(j0 + 32, cur ^ 1);   // overlap DMA with compute
    const _Float16* cK = sK[cur];
    const _Float16* cV = sV[cur];

    // ---- S^T = K @ Q^T : rows j, cols i ----
    v8f S[2];
#pragma unroll
    for (int jj = 0; jj < 2; ++jj) {
      const _Float16* kr = cK + (jj * 16 + col) * ND;
      const v16h ka0 = cat8(*(const v8h*)(kr + kbase),
                            *(const v8h*)(kr + kbase + 8));
      const v16h ka1 = cat8(*(const v8h*)(kr + 32 + kbase),
                            *(const v8h*)(kr + 32 + kbase + 8));
      v8f z = {};
      z = wmma_f16(ka0, qb0, z);
      S[jj] = wmma_f16(ka1, qb1, z);
    }
    // ---- online softmax: lane owns one query; 16 own j's + partner lane ----
    float p0[8], p1[8];
    float mx = -3.0e38f;
#pragma unroll
    for (int r = 0; r < 8; ++r) {
      p0[r] = S[0][r] * scale;
      p1[r] = S[1][r] * scale;
      mx = fmaxf(mx, fmaxf(p0[r], p1[r]));
    }
    mx = fmaxf(mx, __shfl_xor(mx, 16, 32));
    const float mn   = fmaxf(mrun, mx);
    const float corr = __expf(mrun - mn);
    float rs = 0.0f;
#pragma unroll
    for (int r = 0; r < 8; ++r) {
      p0[r] = __expf(p0[r] - mn);
      p1[r] = __expf(p1[r] - mn);
      rs += p0[r] + p1[r];
    }
    rs += __shfl_xor(rs, 16, 32);
    lrun = lrun * corr + rs;
    mrun = mn;
#pragma unroll
    for (int t = 0; t < 4; ++t)
#pragma unroll
      for (int r = 0; r < 8; ++r) acc[t][r] *= corr;

    // ---- P^T C-layout -> B-operand: pack f16 pairs, swap halves via xor16 ---
    int own0[4], own1[4];
#pragma unroll
    for (int q = 0; q < 4; ++q) {
      h2 a, b2;
      a.x  = (_Float16)p0[2 * q]; a.y  = (_Float16)p0[2 * q + 1];
      b2.x = (_Float16)p1[2 * q]; b2.y = (_Float16)p1[2 * q + 1];
      own0[q] = __builtin_bit_cast(int, a);
      own1[q] = __builtin_bit_cast(int, b2);
    }
    v8i pi;
#pragma unroll
    for (int q = 0; q < 4; ++q) {
      const int oth0 = __shfl_xor(own0[q], 16, 32);
      const int oth1 = __shfl_xor(own1[q], 16, 32);
      pi[q]     = hi ? oth1 : own0[q];   // j = kbase+0..7
      pi[q + 4] = hi ? own1[q] : oth0;   // j = kbase+8..15
    }
    const v16h pb = __builtin_bit_cast(v16h, pi);

    // ---- O^T += V^T @ P^T : 4 tiles over d ----
#pragma unroll
    for (int t = 0; t < 4; ++t) {
      const _Float16* vr = cV + (t * 16 + col) * 32;
      const v16h va = cat8(*(const v8h*)(vr + kbase),
                           *(const v8h*)(vr + kbase + 8));
      acc[t] = wmma_f16(va, pb, acc[t]);
    }
    cur ^= 1;
  }

  // ---- epilogue: lane = query column; rows d contiguous -> b128 stores ----
  const int b = bh >> 2, h = bh & 3;
  const float inv = 1.0f / lrun;
  _Float16* orow = obuf + ((size_t)(b * NN + i0 + col)) * NC + h * ND;
#pragma unroll
  for (int t = 0; t < 4; ++t) {
    v8h o;
#pragma unroll
    for (int r = 0; r < 8; ++r) o[r] = (_Float16)(acc[t][r] * inv);
    *(v8h*)(orow + t * 16 + 8 * hi) = o;
  }
}

// ---------------------------------------------------------------------------
// Kernel 3: output projection + residual + transpose back to (B,C,32,32).
// Block = 8 waves = 8 m-tiles of one channel tile; Wo panel in LDS.
// ---------------------------------------------------------------------------
__global__ __launch_bounds__(256)
void out_proj_kernel(const _Float16* __restrict__ obuf,
                     const float* __restrict__ Wo,
                     const float* __restrict__ bo,
                     const float* __restrict__ x,
                     float* __restrict__ out) {
  __shared__ _Float16 sB[16 * 256];              // [col][k]  8 KB
  const int tid  = threadIdx.x;
  const int lane = tid & 31;
  const int wave = tid >> 5;
  const int col  = lane & 15;
  const int hi   = lane >> 4;
  const int nt   = blockIdx.x & 15;
  const int mt   = (blockIdx.x >> 4) * 8 + wave;
  const int col0 = nt * 16;

  {
    const int c  = tid & 15;
    const int k0 = (tid >> 4) * 16;
    v8h t0, t1;
#pragma unroll
    for (int u = 0; u < 8; ++u) {
      t0[u] = (_Float16)Wo[(k0 + u) * NC + col0 + c];
      t1[u] = (_Float16)Wo[(k0 + 8 + u) * NC + col0 + c];
    }
    *(v8h*)(sB + c * 256 + k0)     = t0;
    *(v8h*)(sB + c * 256 + k0 + 8) = t1;
  }
  __syncthreads();

  const int m = mt * 16 + col;
  const int abase = hi ? 8 : 0;
  const int kbase = hi ? 16 : 0;
  const _Float16* orow = obuf + (size_t)m * NC;
  const _Float16* brow = sB + col * 256;

  v8f acc = {};
#pragma unroll
  for (int k0 = 0; k0 < NC; k0 += 32) {
    const v16h afr = cat8(*(const v8h*)(orow + k0 + kbase),
                          *(const v8h*)(orow + k0 + kbase + 8));
    const v16h bfr = cat8(*(const v8h*)(brow + k0 + kbase),
                          *(const v8h*)(brow + k0 + kbase + 8));
    acc = wmma_f16(afr, bfr, acc);
  }

  const float bias = bo[col0 + col];
  const int rowbase = mt * 16 + 8 * hi;          // same b for whole tile
  const int bb  = rowbase >> 10;
  const int nsp = rowbase & (NN - 1);
  const int idx = bb * (NC * NN) + (col0 + col) * NN + nsp;
  const v8f xres = *(const v8f*)(x + idx);       // residual xs[b,n,c]=x[b,c,n]
  v8f o;
#pragma unroll
  for (int r = 0; r < 8; ++r) o[r] = acc[r] + bias + xres[r];
  *(v8f*)(out + idx) = o;
}

// ---------------------------------------------------------------------------
extern "C" void kernel_launch(void* const* d_in, const int* in_sizes, int n_in,
                              void* d_out, int out_size, void* d_ws, size_t ws_size,
                              hipStream_t stream) {
  const float* x  = (const float*)d_in[0];
  const float* Wp = (const float*)d_in[1];
  const float* bp = (const float*)d_in[2];
  const float* Wo = (const float*)d_in[3];
  const float* bo = (const float*)d_in[4];
  float* out = (float*)d_out;

  const size_t elems = (size_t)NB * NH * NN * ND;      // 2M f16 each
  _Float16* qbuf = (_Float16*)d_ws;
  _Float16* kbuf = qbuf + elems;
  _Float16* vbuf = kbuf + elems;
  _Float16* obuf = vbuf + elems;                       // (B,N,H*D) f16
  _Float16* xt   = obuf + elems;                       // (B,N,C)   f16

  xpose_kernel<<<dim3(16, 4, 8), dim3(256), 0, stream>>>(x, xt);
  qkv_proj_kernel<<<dim3(48 * 64), dim3(256), 0, stream>>>(xt, Wp, bp,
                                                           qbuf, kbuf, vbuf);
  attn_kernel<<<dim3(NN / 128, NB * NH), dim3(256), 0, stream>>>(qbuf, kbuf,
                                                                 vbuf, obuf);
  out_proj_kernel<<<dim3(16 * 64), dim3(256), 0, stream>>>(obuf, Wo, bo, x, out);
}
